// obsbot_seq2seq_21088289423669
// MI455X (gfx1250) — matrix operand order repaired and checked by reading, not compiled, not run on hardware
//
#include <hip/hip_runtime.h>
#include <hip/hip_bf16.h>
#include <math.h>

// ---------------- types ----------------
typedef __attribute__((ext_vector_type(4)))  int    i4;
typedef __attribute__((ext_vector_type(8)))  float  v8f;
typedef __attribute__((ext_vector_type(16))) __bf16 v16bf;

union ABPack { i4 q[2]; v16bf v; };

#define IMG   192
#define HW    (IMG*IMG)
#define Bn    8
#define Tt    12
#define Nn    4096           // NPC
#define K3    12288          // 3*NPC
#define N2    8192           // 2*NPC
#define KSPLIT 4
#define KPART (K3/KSPLIT)    // 3072
#define KITER (KPART/32)     // 96

__device__ __forceinline__ unsigned short f2bf(float f) {
    union { float f; unsigned u; } v; v.f = f;
    unsigned r = v.u + 0x7FFFu + ((v.u >> 16) & 1u);   // RNE
    return (unsigned short)(r >> 16);
}

// ---------------- one-time: W[12288,8192] f32 -> Wt[8192,12288] bf16 ----------------
__global__ void k_transpose_w(const float* __restrict__ W, unsigned short* __restrict__ Wt) {
    __shared__ float s[32][33];
    int tx = threadIdx.x;            // 0..31
    int ty = threadIdx.y;            // 0..7
    int bn = blockIdx.x * 32;        // n tile (cols of W)
    int bk = blockIdx.y * 32;        // k tile (rows of W)
    #pragma unroll
    for (int i = 0; i < 4; ++i) {
        int kk = ty + i * 8;
        s[kk][tx] = W[(size_t)(bk + kk) * N2 + bn + tx];
    }
    __syncthreads();
    #pragma unroll
    for (int i = 0; i < 4; ++i) {
        int nn = ty + i * 8;
        Wt[(size_t)(bn + nn) * K3 + bk + tx] = f2bf(s[tx][nn]);
    }
}

// ---------------- one-time per launch: init XY to uniform grid ----------------
__global__ void k_init_xy(float* __restrict__ XY) {
    int i = blockIdx.x * blockDim.x + threadIdx.x;   // 0 .. 8*2*4096-1
    if (i >= Bn * 2 * Nn) return;
    int n = i % Nn;
    int d = (i / Nn) % 2;
    float v = (d == 0) ? (float)(n % 64) * (1.0f / 63.0f)
                       : (float)(n / 64) * (1.0f / 63.0f);
    XY[i] = v;
}

// ---------------- one-time per launch: zero A rows 8..15 ----------------
__global__ void k_zero_arows(unsigned short* __restrict__ A) {
    int i = blockIdx.x * blockDim.x + threadIdx.x;   // 0 .. 8*12288-1
    if (i < 8 * K3) A[8 * K3 + i] = 0;
}

// ---------------- per step: bilinear sample + pack A (bf16) ----------------
__global__ void k_sample_pack(const float* __restrict__ inp, int t,
                              const float* __restrict__ XY,
                              float* __restrict__ Rpc,
                              unsigned short* __restrict__ A) {
    int i = blockIdx.x * blockDim.x + threadIdx.x;   // 0 .. 8*4096-1
    if (i >= Bn * Nn) return;
    int b = i / Nn, n = i % Nn;
    float x = XY[(size_t)b * 2 * Nn + n];
    float y = XY[(size_t)b * 2 * Nn + Nn + n];
    float fx = x * (float)IMG - 0.5f;
    float fy = y * (float)IMG - 0.5f;
    float x0 = floorf(fx), y0 = floorf(fy);
    float wx = fx - x0, wy = fy - y0;
    const float* img = inp + ((size_t)b * Tt + t) * HW;
    int ix0 = (int)x0, iy0 = (int)y0;
    auto tap = [&](int ix, int iy) -> float {
        if (ix < 0 || ix > IMG - 1 || iy < 0 || iy > IMG - 1) return 0.0f;
        return img[iy * IMG + ix];
    };
    float v = tap(ix0,     iy0)     * (1.0f - wx) * (1.0f - wy)
            + tap(ix0 + 1, iy0)     * wx          * (1.0f - wy)
            + tap(ix0,     iy0 + 1) * (1.0f - wx) * wy
            + tap(ix0 + 1, iy0 + 1) * wx          * wy;
    Rpc[i] = v;
    unsigned short* Ar = A + (size_t)b * K3;
    Ar[n]            = f2bf(x);
    Ar[Nn + n]       = f2bf(y);
    Ar[2 * Nn + n]   = f2bf(v);
}

// ---------------- per step: the WMMA GEMM  [16 x 12288] x [12288 x 8192] ----------------
// 2048 waves: wave = (ntile 0..511, kpart 0..3); partial sums P[4][8][8192] f32.
__global__ __launch_bounds__(256) void k_gemm(const unsigned short* __restrict__ A,
                                              const unsigned short* __restrict__ Wt,
                                              float* __restrict__ P) {
    int lane = threadIdx.x & 31;
    int wid  = (blockIdx.x << 3) + (threadIdx.x >> 5);  // 0..2047
    int ntile = wid & 511;
    int kp    = wid >> 9;
    int col16 = lane & 15;
    int nc    = (ntile << 4) + col16;
    // A layout (16-bit A 16x32): lane row M = lane%16; K-base 0 (lanes 0-15) / 8 (lanes 16-31)
    const unsigned short* Ar = A  + (size_t)col16 * K3 + ((lane < 16) ? 0 : 8);
    // B layout (16-bit B 32x16): lane col N = lane%16; K-base 0 (lanes 0-15) / 8 (lanes 16-31), same as A
    const unsigned short* Br = Wt + (size_t)nc    * K3 + ((lane < 16) ? 0 : 8);
    int kstart = kp * KPART;
    v8f acc = {};
    #pragma unroll 2
    for (int kk = 0; kk < KITER; ++kk) {
        int k0 = kstart + (kk << 5);
        __builtin_prefetch(Br + k0 + 2048, 0, 0);      // global_prefetch_b8, ~4KB ahead on W stream
        ABPack a, bmat;
        a.q[0]    = *(const i4*)(Ar + k0);             // VGPR0-3: K = kbase+0..7
        a.q[1]    = *(const i4*)(Ar + k0 + 16);        // VGPR4-7: K = 16+kbase+0..7
        bmat.q[0] = *(const i4*)(Br + k0);             // K = kbase+0..7
        bmat.q[1] = *(const i4*)(Br + k0 + 16);        // K = 16+kbase+0..7
        acc = __builtin_amdgcn_wmma_f32_16x16x32_bf16(
                false, a.v, false, bmat.v, (short)0, acc, false, false);
    }
    // C/D layout: lane L, VGPR r -> (M = r + (L>=16?8:0), N = L%16). Rows 0..7 are the batch.
    if (lane < 16) {
        float* dst = P + (size_t)(kp * 8) * N2 + nc;
        #pragma unroll
        for (int r = 0; r < 8; ++r) dst[(size_t)r * N2] = acc[r];
    }
}

// ---------------- per step: bias + tanh + XY update ----------------
__global__ void k_finalize(const float* __restrict__ P, const float* __restrict__ bias,
                           float* __restrict__ XY) {
    int i = blockIdx.x * blockDim.x + threadIdx.x;   // 0 .. 8*8192-1
    if (i >= Bn * N2) return;
    int b = i / N2, j = i % N2;
    float pre = bias[j];
    #pragma unroll
    for (int kp = 0; kp < KSPLIT; ++kp) pre += P[(size_t)(kp * 8 + b) * N2 + j];
    float d = 0.01f * tanhf(pre);
    if (j < Nn) XY[(size_t)b * 2 * Nn + j]             += d;
    else        XY[(size_t)b * 2 * Nn + Nn + (j - Nn)] += d;
}

// ---------------- per step: zero output slice + weight accumulator ----------------
__global__ void k_zero_grid(float* __restrict__ out, int t, float* __restrict__ wsum) {
    int i = blockIdx.x * blockDim.x + threadIdx.x;   // 0 .. 8*HW-1
    if (i >= Bn * HW) return;
    int b = i / HW, p = i % HW;
    out[((size_t)b * Tt + t) * HW + p] = 0.0f;
    wsum[i] = 0.0f;
}

// ---------------- per step: reverse-bilinear splat (fp32 atomics) ----------------
__global__ void k_splat(const float* __restrict__ Rpc, const float* __restrict__ XY,
                        float* __restrict__ out, int t, float* __restrict__ wsum) {
    int i = blockIdx.x * blockDim.x + threadIdx.x;   // 0 .. 8*4096-1
    if (i >= Bn * Nn) return;
    int b = i / Nn, n = i % Nn;
    float v  = Rpc[i];
    float x  = XY[(size_t)b * 2 * Nn + n];
    float y  = XY[(size_t)b * 2 * Nn + Nn + n];
    float fx = x * (float)IMG - 0.5f;
    float fy = y * (float)IMG - 0.5f;
    float x0 = floorf(fx), y0 = floorf(fy);
    float wx = fx - x0, wy = fy - y0;
    int ix0 = (int)x0, iy0 = (int)y0;
    float* num = out + ((size_t)b * Tt + t) * HW;
    float* wsb = wsum + (size_t)b * HW;
    #pragma unroll
    for (int c = 0; c < 4; ++c) {
        int dx = c & 1, dy = c >> 1;
        int ix = ix0 + dx, iy = iy0 + dy;
        if (ix < 0 || ix > IMG - 1 || iy < 0 || iy > IMG - 1) continue;
        float w = (dx ? wx : 1.0f - wx) * (dy ? wy : 1.0f - wy);
        atomicAdd(&num[iy * IMG + ix], v * w);
        atomicAdd(&wsb[iy * IMG + ix], w);
    }
}

// ---------------- per step: normalize ----------------
__global__ void k_normalize(float* __restrict__ out, int t, const float* __restrict__ wsum) {
    int i = blockIdx.x * blockDim.x + threadIdx.x;
    if (i >= Bn * HW) return;
    int b = i / HW, p = i % HW;
    size_t o = ((size_t)b * Tt + t) * HW + p;
    out[o] = out[o] / (wsum[i] + 1e-10f);
}

// ---------------- host ----------------
extern "C" void kernel_launch(void* const* d_in, const int* in_sizes, int n_in,
                              void* d_out, int out_size, void* d_ws, size_t ws_size,
                              hipStream_t stream) {
    const float* inp  = (const float*)d_in[0];   // [8,12,1,192,192]
    const float* Wl   = (const float*)d_in[1];   // [12288,8192]
    const float* bl   = (const float*)d_in[2];   // [8192]
    float* out = (float*)d_out;

    char* ws = (char*)d_ws;
    size_t offWt = 0;                                        // bf16 W^T  : 8192*12288*2
    size_t offA  = offWt + (size_t)N2 * K3 * 2;              // bf16 A    : 16*12288*2
    size_t offXY = offA  + (size_t)16 * K3 * 2;              // f32 XY    : 8*2*4096
    size_t offR  = offXY + (size_t)Bn * 2 * Nn * 4;          // f32 R_pc  : 8*4096
    size_t offP  = offR  + (size_t)Bn * Nn * 4;              // f32 P     : 4*8*8192
    size_t offWs = offP  + (size_t)KSPLIT * 8 * N2 * 4;      // f32 wsum  : 8*HW

    unsigned short* Wt = (unsigned short*)(ws + offWt);
    unsigned short* A  = (unsigned short*)(ws + offA);
    float* XY   = (float*)(ws + offXY);
    float* Rpc  = (float*)(ws + offR);
    float* P    = (float*)(ws + offP);
    float* wsum = (float*)(ws + offWs);

    // one-time per launch
    k_transpose_w<<<dim3(N2 / 32, K3 / 32), dim3(32, 8), 0, stream>>>(Wl, Wt);
    k_init_xy   <<<(Bn * 2 * Nn + 255) / 256, 256, 0, stream>>>(XY);
    k_zero_arows<<<(8 * K3 + 255) / 256, 256, 0, stream>>>(A);

    for (int t = 0; t < Tt; ++t) {
        k_sample_pack<<<(Bn * Nn + 255) / 256, 256, 0, stream>>>(inp, t, XY, Rpc, A);
        k_gemm       <<<(512 * KSPLIT) / 8, 256, 0, stream>>>(A, Wt, P);
        k_finalize   <<<(Bn * N2 + 255) / 256, 256, 0, stream>>>(P, bl, XY);
        k_zero_grid  <<<(Bn * HW + 255) / 256, 256, 0, stream>>>(out, t, wsum);
        k_splat      <<<(Bn * Nn + 255) / 256, 256, 0, stream>>>(Rpc, XY, out, t, wsum);
        k_normalize  <<<(Bn * HW + 255) / 256, 256, 0, stream>>>(out, t, wsum);
    }
    (void)in_sizes; (void)n_in; (void)out_size; (void)ws_size;
}
